// Graphormer3DGraphEncoderLayer_67250597921316
// MI455X (gfx1250) — matrix operand order repaired and checked by reading, not compiled, not run on hardware
//
#include <hip/hip_runtime.h>
#include <hip/hip_bf16.h>

// ---------------------------------------------------------------------------
// Graphormer encoder layer for MI455X (gfx1250, wave32, WMMA).
// All GEMMs use v_wmma_f32_16x16x32_bf16 (f32 accum) with double-buffered LDS
// tiles filled by GLOBAL_LOAD_ASYNC_TO_LDS_B128 (ASYNCcnt path, no VGPR
// staging, loads overlap WMMA compute). Attention is fused (bias + softmax +
// P@V) per (batch, 64-row block) in ~204KB dynamic LDS so the 384x512x512
// score/prob tensors never touch HBM; the unavoidable 402MB attn_bias read
// dominates traffic (~17us at 23.3 TB/s).
// Workspace requirement: ~294 MB (regions are time-multiplexed, see offsets).
// ---------------------------------------------------------------------------

typedef unsigned short u16;
typedef unsigned int   u32;
typedef __attribute__((ext_vector_type(16))) __bf16 v16bf;
typedef __attribute__((ext_vector_type(8)))  float  v8f;

#define N_NODE   512
#define N_GRAPH  32
#define EMBED    768
#define HEADS    12
#define HEAD_DIM 64
#define FFN_DIM  3072
#define ROWS     (N_NODE * N_GRAPH)   /* 16384 */
#define BATCHES  (N_GRAPH * HEADS)    /* 384   */

struct alignas(16) U4 { u32 a, b, c, d; };   // trivial 16B chunk (safe in unions)

__device__ __forceinline__ u16 f2bf(float f) {
  u32 u = __float_as_uint(f);
  u32 r = u + 0x7FFFu + ((u >> 16) & 1u);    // round-to-nearest-even
  return (u16)(r >> 16);
}

__device__ __forceinline__ v16bf make_frag(U4 lo, U4 hi) {
  union { U4 q[2]; v16bf v; } un;
  un.q[0] = lo; un.q[1] = hi;
  return un.v;
}

__device__ __forceinline__ v8f wmma_bf16(v16bf a, v16bf b, v8f c) {
  // D = A(16x32 bf16) x B(32x16 bf16) + C(16x16 f32)
  return __builtin_amdgcn_wmma_f32_16x16x32_bf16(false, a, false, b,
                                                 (short)0, c, false, false);
}

// 32-bit LDS byte offset of a __shared__ object (for async-to-LDS dest VGPR)
__device__ __forceinline__ u32 lds_off_of(const void* p) {
  return (u32)(size_t)(const __attribute__((address_space(3))) void*)p;
}

// GLOBAL_LOAD_ASYNC_TO_LDS_B128: copies 16 bytes from global memory to LDS, ASYNCcnt-tracked.
__device__ __forceinline__ void async_b128(u32 lds_byte_off, const void* gaddr) {
  asm volatile("global_load_async_to_lds_b128 %0, %1, off"
               :: "v"(lds_byte_off), "v"(gaddr) : "memory");
}

__device__ __forceinline__ void wait_async0() {
  asm volatile("s_wait_asynccnt 0x0" ::: "memory");
}

// ---------------------------------------------------------------------------
// f32 -> bf16 elementwise cast (weights)
// ---------------------------------------------------------------------------
__global__ __launch_bounds__(256) void cast_to_bf16(const float* __restrict__ x,
                                                    u16* __restrict__ y, int n) {
  int i = blockIdx.x * 256 + threadIdx.x;
  if (i < n) y[i] = f2bf(x[i]);
}

// ---------------------------------------------------------------------------
// LayerNorm over last dim (768) + cast to bf16. One block (256 thr) per row.
// ---------------------------------------------------------------------------
__global__ __launch_bounds__(256) void ln_to_bf16(const float* __restrict__ X,
                                                  const float* __restrict__ gamma,
                                                  const float* __restrict__ beta,
                                                  u16* __restrict__ Y) {
  __shared__ float sm[8];
  __shared__ float stat[2];
  const int t = threadIdx.x;
  const size_t base = (size_t)blockIdx.x * EMBED;
  const float x0 = X[base + t];
  const float x1 = X[base + t + 256];
  const float x2 = X[base + t + 512];
  float s = x0 + x1 + x2;
#pragma unroll
  for (int o = 16; o > 0; o >>= 1) s += __shfl_xor(s, o, 32);
  if ((t & 31) == 0) sm[t >> 5] = s;
  __syncthreads();
  if (t == 0) {
    float tt = 0.f;
    for (int i = 0; i < 8; ++i) tt += sm[i];
    stat[0] = tt * (1.0f / EMBED);
  }
  __syncthreads();
  const float mean = stat[0];
  const float d0 = x0 - mean, d1 = x1 - mean, d2 = x2 - mean;
  float v = d0 * d0 + d1 * d1 + d2 * d2;
#pragma unroll
  for (int o = 16; o > 0; o >>= 1) v += __shfl_xor(v, o, 32);
  __syncthreads();
  if ((t & 31) == 0) sm[t >> 5] = v;
  __syncthreads();
  if (t == 0) {
    float tt = 0.f;
    for (int i = 0; i < 8; ++i) tt += sm[i];
    stat[1] = tt * (1.0f / EMBED);
  }
  __syncthreads();
  const float rstd = rsqrtf(stat[1] + 1e-5f);
  Y[base + t]       = f2bf(d0 * rstd * gamma[t]       + beta[t]);
  Y[base + t + 256] = f2bf(d1 * rstd * gamma[t + 256] + beta[t + 256]);
  Y[base + t + 512] = f2bf(d2 * rstd * gamma[t + 512] + beta[t + 512]);
}

// ---------------------------------------------------------------------------
// Generic WMMA GEMM: C[M,N] = A[M,K](bf16) @ B[N,K]^T(bf16) + bias[N]
//   EPI 0: store f32
//   EPI 1: store f32, += resid[M,N]
//   EPI 2: exact GELU, store bf16
// 128x128 block tile, 256 threads = 8 waves (2x4 wave grid, 64x32 per wave).
// K staged in 32-wide bf16 LDS tiles (row stride 40 to spread banks),
// double-buffered, filled with async-to-LDS loads overlapping the WMMAs.
// ---------------------------------------------------------------------------
template <int EPI>
__global__ __launch_bounds__(256) void gemm_bf16(const u16* __restrict__ A,
                                                 const u16* __restrict__ B,
                                                 const float* __restrict__ bias,
                                                 const float* __restrict__ resid,
                                                 void* __restrict__ Cout,
                                                 int M, int N, int K) {
  __shared__ alignas(16) u16 As[2][128 * 40];
  __shared__ alignas(16) u16 Bs[2][128 * 40];
  const int t    = threadIdx.x;
  const int lane = t & 31;
  const int wave = t >> 5;
  const int wr   = wave >> 2;              // 0..1 (row group of 64)
  const int wc   = wave & 3;               // 0..3 (col group of 32)
  const int m0   = blockIdx.y * 128;
  const int n0   = blockIdx.x * 128;
  const int h8   = (lane >> 4) * 8;        // A-frag K-chunk select
  const int k16  = (lane >> 4) * 16;       // B-frag offset, cancelled below: B also uses h8 K-chunks

  v8f acc[4][2] = {};

  // staging: thread t owns row t/2, 16-bf16 half (t&1) of the 32-wide K slice
  const int srow = t >> 1;
  const int skh  = (t & 1) * 16;
  const u16* gA = A + (size_t)(m0 + srow) * K + skh;
  const u16* gB = B + (size_t)(n0 + srow) * K + skh;
  u32 lA[2], lB[2];
  lA[0] = lds_off_of(&As[0][srow * 40 + skh]);
  lA[1] = lds_off_of(&As[1][srow * 40 + skh]);
  lB[0] = lds_off_of(&Bs[0][srow * 40 + skh]);
  lB[1] = lds_off_of(&Bs[1][srow * 40 + skh]);

  // prologue: fill buffer 0
  async_b128(lA[0], gA);      async_b128(lA[0] + 16, gA + 8);
  async_b128(lB[0], gB);      async_b128(lB[0] + 16, gB + 8);
  wait_async0();
  __syncthreads();

  int cur = 0;
  for (int k0 = 0; k0 < K; k0 += 32) {
    if (k0 + 32 < K) {            // prefetch next K-tile into the other buffer
      const u16* pa = gA + k0 + 32;
      const u16* pb = gB + k0 + 32;
      const int nxt = cur ^ 1;
      async_b128(lA[nxt], pa);      async_b128(lA[nxt] + 16, pa + 8);
      async_b128(lB[nxt], pb);      async_b128(lB[nxt] + 16, pb + 8);
    }
    v16bf af[4], bfr[2];
#pragma unroll
    for (int i = 0; i < 4; ++i) {
      const u16* p = &As[cur][(wr * 64 + i * 16 + (lane & 15)) * 40];
      af[i] = make_frag(*(const U4*)(p + h8), *(const U4*)(p + 16 + h8));
    }
#pragma unroll
    for (int j = 0; j < 2; ++j) {
      const u16* p = &Bs[cur][(wc * 32 + j * 16 + (lane & 15)) * 40 + k16];
      bfr[j] = make_frag(*(const U4*)(p - k16 + h8), *(const U4*)(p - k16 + 16 + h8));
    }
#pragma unroll
    for (int i = 0; i < 4; ++i)
#pragma unroll
      for (int j = 0; j < 2; ++j)
        acc[i][j] = wmma_bf16(af[i], bfr[j], acc[i][j]);
    wait_async0();                // next tile landed (overlapped with WMMAs)
    __syncthreads();
    cur ^= 1;
  }

  const int rbase = (lane >> 4) * 8;
#pragma unroll
  for (int i = 0; i < 4; ++i) {
#pragma unroll
    for (int j = 0; j < 2; ++j) {
      const int gcol = n0 + wc * 32 + j * 16 + (lane & 15);
      const float bv = bias[gcol];
#pragma unroll
      for (int v = 0; v < 8; ++v) {
        const int gm = m0 + wr * 64 + i * 16 + rbase + v;
        float val = acc[i][j][v] + bv;
        const size_t idx = (size_t)gm * N + gcol;
        if (EPI == 1) {
          ((float*)Cout)[idx] = val + resid[idx];
        } else if (EPI == 2) {
          val = 0.5f * val * (1.0f + erff(val * 0.70710678118f));
          ((u16*)Cout)[idx] = f2bf(val);
        } else {
          ((float*)Cout)[idx] = val;
        }
      }
    }
  }
}

// ---------------------------------------------------------------------------
// Split QKV f32 [16384, 2304] -> head-major bf16 q(scaled)/k/v [384,512,64]
// ---------------------------------------------------------------------------
__global__ __launch_bounds__(256) void split_qkv(const float* __restrict__ qkv,
                                                 u16* __restrict__ q,
                                                 u16* __restrict__ k,
                                                 u16* __restrict__ v) {
  const size_t total = (size_t)ROWS * 3 * EMBED;
  for (size_t i = (size_t)blockIdx.x * blockDim.x + threadIdx.x; i < total;
       i += (size_t)gridDim.x * blockDim.x) {
    const int row = (int)(i / (3 * EMBED));
    const int col = (int)(i % (3 * EMBED));
    const int n = row / N_GRAPH;
    const int g = row % N_GRAPH;
    const int which = col / EMBED;
    const int c = col % EMBED;
    const int h = c / HEAD_DIM;
    const int d = c % HEAD_DIM;
    const size_t oi = (((size_t)(g * HEADS + h)) * N_NODE + n) * HEAD_DIM + d;
    const float val = qkv[i];
    if (which == 0)      q[oi] = f2bf(val * 0.125f);   // d^-0.5, d=64
    else if (which == 1) k[oi] = f2bf(val);
    else                 v[oi] = f2bf(val);
  }
}

// ---------------------------------------------------------------------------
// Fused attention: per (batch b, 64-row block): S = qK^T (+bias), softmax,
// O = P@V, all through LDS. 128 threads = 4 waves, wave w owns rows w*16..+15.
// Dynamic LDS: S f32 [64][516] | P bf16 [64][520] | Vt bf16 [64][72] | red|lrow
// ---------------------------------------------------------------------------
#define S_STRIDE  516
#define P_STRIDE  520
#define VT_STRIDE 72
#define S_BYTES   (64 * S_STRIDE * 4)
#define P_BYTES   (64 * P_STRIDE * 2)
#define VT_BYTES  (64 * VT_STRIDE * 2)
#define ATTN_SMEM (S_BYTES + P_BYTES + VT_BYTES + 128 * 4 + 64 * 4)

__global__ __launch_bounds__(128) void attn_kernel(const u16* __restrict__ Q,
                                                   const u16* __restrict__ Km,
                                                   const u16* __restrict__ V,
                                                   const float* __restrict__ bias,
                                                   u16* __restrict__ ctx) {
  extern __shared__ char smem[];
  float* S    = (float*)smem;
  u16*   P    = (u16*)(smem + S_BYTES);
  u16*   Vt   = (u16*)(smem + S_BYTES + P_BYTES);
  float* red  = (float*)(smem + S_BYTES + P_BYTES + VT_BYTES);
  float* lrow = red + 128;

  const int t    = threadIdx.x;
  const int lane = t & 31;
  const int wave = t >> 5;
  const int b    = blockIdx.y;            // batch = g*HEADS + h
  const int row0 = blockIdx.x * 64;
  const int h8   = (lane >> 4) * 8;
  const int k16  = (lane >> 4) * 16;

  // ---- Phase 0: Q A-fragments (d=64 -> two 32-wide frags), kept in VGPRs ----
  v16bf qf[2];
  {
    const u16* qrow =
        Q + ((size_t)b * N_NODE + row0 + wave * 16 + (lane & 15)) * HEAD_DIM;
    qf[0] = make_frag(*(const U4*)(qrow + h8),      *(const U4*)(qrow + 16 + h8));
    qf[1] = make_frag(*(const U4*)(qrow + 32 + h8), *(const U4*)(qrow + 48 + h8));
  }

  // ---- Phase 1: S[64][512] = q @ K^T into LDS (32 col-tiles of 16) ----
  for (int ct = 0; ct < 32; ++ct) {
    const u16* krow =
        Km + ((size_t)b * N_NODE + ct * 16 + (lane & 15)) * HEAD_DIM;
    v16bf b0 = make_frag(*(const U4*)(krow + h8),      *(const U4*)(krow + 16 + h8));
    v16bf b1 = make_frag(*(const U4*)(krow + 32 + h8), *(const U4*)(krow + 48 + h8));
    v8f c = {};
    c = wmma_bf16(qf[0], b0, c);
    c = wmma_bf16(qf[1], b1, c);
#pragma unroll
    for (int v = 0; v < 8; ++v)
      S[(wave * 16 + h8 + v) * S_STRIDE + ct * 16 + (lane & 15)] = c[v];
  }
  __syncthreads();

  // ---- Phase 2: += bias, softmax (unnormalized P bf16, row sums -> lrow) ----
  {
    const int row   = t >> 1;
    const int cbase = (t & 1) * 256;
    const float* brow =
        bias + ((size_t)b * N_NODE + row0 + row) * N_NODE + cbase;
    float* srow = S + row * S_STRIDE + cbase;
    float mx = -3.0e38f;
    for (int c = 0; c < 256; ++c) {
      float s = srow[c] + brow[c];
      srow[c] = s;
      mx = fmaxf(mx, s);
    }
    red[t] = mx;
    __syncthreads();
    const float m = fmaxf(red[row * 2], red[row * 2 + 1]);
    __syncthreads();
    float sum = 0.0f;
    u16* prow = P + row * P_STRIDE + cbase;
    for (int c = 0; c < 256; ++c) {
      float e = __expf(srow[c] - m);
      prow[c] = f2bf(e);
      sum += e;
    }
    red[t] = sum;
    __syncthreads();
    if ((t & 1) == 0) lrow[row] = red[row * 2] + red[row * 2 + 1];
  }
  __syncthreads();

  // ---- Phase 3: O = P @ V, streaming V in 64-row chunks (LDS-transposed) ----
  v8f O[4] = {};
  for (int chunk = 0; chunk < 8; ++chunk) {
    {
      const int ml = t >> 1;
      const int d0 = (t & 1) * 32;
      const u16* vsrc =
          V + ((size_t)b * N_NODE + chunk * 64 + ml) * HEAD_DIM + d0;
      alignas(16) u16 tmp[32];
      ((U4*)tmp)[0] = ((const U4*)vsrc)[0];
      ((U4*)tmp)[1] = ((const U4*)vsrc)[1];
      ((U4*)tmp)[2] = ((const U4*)vsrc)[2];
      ((U4*)tmp)[3] = ((const U4*)vsrc)[3];
#pragma unroll
      for (int j = 0; j < 32; ++j)
        Vt[(d0 + j) * VT_STRIDE + ml] = tmp[j];   // Vt[d][m]
    }
    __syncthreads();
#pragma unroll
    for (int kk = 0; kk < 2; ++kk) {
      const u16* prow =
          P + (wave * 16 + (lane & 15)) * P_STRIDE + chunk * 64 + kk * 32;
      v16bf a = make_frag(*(const U4*)(prow + h8), *(const U4*)(prow + 16 + h8));
#pragma unroll
      for (int nt = 0; nt < 4; ++nt) {
        const u16* vrow = Vt + (nt * 16 + (lane & 15)) * VT_STRIDE + kk * 32 + k16;
        v16bf bv = make_frag(*(const U4*)(vrow - k16 + h8), *(const U4*)(vrow - k16 + 16 + h8));
        O[nt] = wmma_bf16(a, bv, O[nt]);
      }
    }
    __syncthreads();
  }

  // ---- Phase 4: normalize rows, scatter bf16 into context [n, g, h*64+d] ----
  const int g = b / HEADS;
  const int h = b % HEADS;
#pragma unroll
  for (int nt = 0; nt < 4; ++nt) {
#pragma unroll
    for (int v = 0; v < 8; ++v) {
      const int r = wave * 16 + h8 + v;
      const float val = O[nt][v] / lrow[r];
      const int n = row0 + r;
      ctx[((size_t)n * N_GRAPH + g) * EMBED + h * HEAD_DIM + nt * 16 +
          (lane & 15)] = f2bf(val);
    }
  }
}

// ---------------------------------------------------------------------------
// Workspace layout (bytes). Peak requirement ~294 MB; regions time-multiplexed.
// ---------------------------------------------------------------------------
static const size_t OFF_WIN  = 0;            // w_in  bf16  3,538,944
static const size_t OFF_WOUT = 3538944;      // w_out bf16  1,179,648
static const size_t OFF_FC1  = 4718592;      // fc1_w bf16  4,718,592
static const size_t OFF_FC2  = 9437184;      // fc2_w bf16  4,718,592
static const size_t OFF_XLN  = 16777216;     // x_ln bf16 25.2MB; later ctx bf16
static const size_t OFF_QKV  = 50331648;     // qkv f32 151MB; later h1 bf16 101MB
static const size_t OFF_X    = 159383552;    // residual-2 f32 50.3MB
static const size_t OFF_Q    = 218103808;    // q bf16 25.2MB; later y_ln bf16
static const size_t OFF_K    = 243269632;    // k bf16 25.2MB
static const size_t OFF_V    = 268435456;    // v bf16 25.2MB (end ~293.6MB)

extern "C" void kernel_launch(void* const* d_in, const int* in_sizes, int n_in,
                              void* d_out, int out_size, void* d_ws,
                              size_t ws_size, hipStream_t stream) {
  (void)in_sizes; (void)n_in; (void)out_size; (void)ws_size;
  const float* input_nodes = (const float*)d_in[0];
  const float* attn_bias   = (const float*)d_in[1];
  const float* w_in        = (const float*)d_in[2];
  const float* b_in        = (const float*)d_in[3];
  const float* w_out       = (const float*)d_in[4];
  const float* b_out       = (const float*)d_in[5];
  const float* ln1_s       = (const float*)d_in[6];
  const float* ln1_b       = (const float*)d_in[7];
  const float* fc1_w       = (const float*)d_in[8];
  const float* fc1_b       = (const float*)d_in[9];
  const float* fc2_w       = (const float*)d_in[10];
  const float* fc2_b       = (const float*)d_in[11];
  const float* ln2_s       = (const float*)d_in[12];
  const float* ln2_b       = (const float*)d_in[13];

  char* ws = (char*)d_ws;
  u16*   w_in_bf  = (u16*)(ws + OFF_WIN);
  u16*   w_out_bf = (u16*)(ws + OFF_WOUT);
  u16*   fc1_bf   = (u16*)(ws + OFF_FC1);
  u16*   fc2_bf   = (u16*)(ws + OFF_FC2);
  u16*   xln      = (u16*)(ws + OFF_XLN);
  u16*   ctx      = (u16*)(ws + OFF_XLN);    // reuse: x_ln dead after QKV GEMM
  float* qkvf     = (float*)(ws + OFF_QKV);
  u16*   h1       = (u16*)(ws + OFF_QKV);    // reuse: qkv dead after split
  float* xres     = (float*)(ws + OFF_X);
  u16*   qbf      = (u16*)(ws + OFF_Q);
  u16*   yln      = (u16*)(ws + OFF_Q);      // reuse: q dead after attention
  u16*   kbf      = (u16*)(ws + OFF_K);
  u16*   vbf      = (u16*)(ws + OFF_V);

  // 1. weights -> bf16
  cast_to_bf16<<<(2304 * 768 + 255) / 256, 256, 0, stream>>>(w_in, w_in_bf, 2304 * 768);
  cast_to_bf16<<<(768 * 768 + 255) / 256, 256, 0, stream>>>(w_out, w_out_bf, 768 * 768);
  cast_to_bf16<<<(3072 * 768 + 255) / 256, 256, 0, stream>>>(fc1_w, fc1_bf, 3072 * 768);
  cast_to_bf16<<<(3072 * 768 + 255) / 256, 256, 0, stream>>>(fc2_w, fc2_bf, 3072 * 768);

  // 2. LN1 -> bf16
  ln_to_bf16<<<ROWS, 256, 0, stream>>>(input_nodes, ln1_s, ln1_b, xln);

  // 3. QKV GEMM: [16384,2304] = xln @ w_in^T + b_in   (f32 out)
  gemm_bf16<0><<<dim3(3 * EMBED / 128, ROWS / 128), 256, 0, stream>>>(
      xln, w_in_bf, b_in, nullptr, qkvf, ROWS, 3 * EMBED, EMBED);

  // 4. split to head-major q(scaled)/k/v bf16
  split_qkv<<<147456, 256, 0, stream>>>(qkvf, qbf, kbf, vbf);

  // 5. fused attention -> context bf16
  (void)hipFuncSetAttribute(reinterpret_cast<const void*>(attn_kernel),
                            hipFuncAttributeMaxDynamicSharedMemorySize,
                            ATTN_SMEM);
  attn_kernel<<<dim3(N_NODE / 64, BATCHES), 128, ATTN_SMEM, stream>>>(
      qbf, kbf, vbf, attn_bias, ctx);

  // 6. out-proj + residual: xres = input + ctx @ w_out^T + b_out  (f32)
  gemm_bf16<1><<<dim3(EMBED / 128, ROWS / 128), 256, 0, stream>>>(
      ctx, w_out_bf, b_out, input_nodes, xres, ROWS, EMBED, EMBED);

  // 7. LN2 -> bf16
  ln_to_bf16<<<ROWS, 256, 0, stream>>>(xres, ln2_s, ln2_b, yln);

  // 8. FC1 + exact GELU -> h1 bf16 [16384,3072]
  gemm_bf16<2><<<dim3(FFN_DIM / 128, ROWS / 128), 256, 0, stream>>>(
      yln, fc1_bf, fc1_b, nullptr, h1, ROWS, FFN_DIM, EMBED);

  // 9. FC2 + residual -> d_out f32 [16384,768]
  gemm_bf16<1><<<dim3(EMBED / 128, ROWS / 128), 256, 0, stream>>>(
      h1, fc2_bf, fc2_b, xres, d_out, ROWS, EMBED, FFN_DIM);
}
